// iNGPDW_57836029607991
// MI455X (gfx1250) — matrix operand order repaired and checked by reading, not compiled, not run on hardware
//
#include <hip/hip_runtime.h>
#include <hip/hip_fp16.h>

typedef __attribute__((ext_vector_type(16))) _Float16 v16h;
typedef __attribute__((ext_vector_type(8)))  float    v8f;
typedef __attribute__((ext_vector_type(4)))  float    vf4;

#define LVLS 10
#define TSIZE 65536

struct HeadW {
  const float* W1; const float* b1;
  const float* W2; const float* b2;
  const float* W3; const float* b3;
};

struct Params {
  const float* x;
  const float* cr;
  const float* tables;
  HeadW h[5];
  float* out;
  int n;
};

// A-matrix (16x32 f16) element placement per CDNA5 ISA 7.12.2:
// lanes 0-15: row=lane, K in {0..7 -> slots 0..7, 16..23 -> slots 8..15}
// lanes 16-31: row=lane-16, K in {8..15 -> slots 0..7, 24..31 -> slots 8..15}
__device__ __forceinline__ void store_a_frag(_Float16* base, int row, int k, _Float16 v) {
  int c    = k >> 5;
  int kk   = k & 31;
  int grp  = (kk >> 3) & 1;
  int slot = (kk & 7) + ((kk >> 4) << 3);
  int lane = row + grp * 16;
  base[(c * 32 + lane) * 16 + slot] = v;
}

// B-matrix (32x16 f16): col = lane&15; lanes 0-15 hold K in {0..7, 16..23},
// lanes 16-31 hold K in {8..15, 24..31}  (same K order as A-frag)
__device__ __forceinline__ void store_b_frag(_Float16* base, int fragBase, int k, int n, _Float16 v) {
  int c    = k >> 5;
  int kk   = k & 31;
  int grp  = (kk >> 3) & 1;
  int slot = (kk & 7) + ((kk >> 4) << 3);
  int lane = (n & 15) + grp * 16;
  int fi   = fragBase + (n >> 4) * 2 + c;
  base[(fi * 32 + lane) * 16 + slot] = v;
}

// Branchless SELU: exp is always executed on min(x,0) so the compiler emits
// v_cndmask instead of exec-mask branches around v_exp_f32.
__device__ __forceinline__ float selu_f(float x) {
  const float a = 1.6732632423543772f;
  const float s = 1.0507009873554805f;
  float e = __expf(fminf(x, 0.f)) - 1.f;
  return x > 0.f ? s * x : s * a * e;
}

__launch_bounds__(128)
__global__ void ingp_fused_wmma(Params P) {
  __shared__ v16h  hFrag[8 * 2 * 32];   // encoded features, A-frag layout, K padded to 64
  __shared__ v16h  actFrag[8 * 2 * 32]; // hidden activations, A-frag layout
  __shared__ v16h  bFrag[18 * 32];      // current head weights: L1 frags 0-7, L2 8-15, L3 16-17
  __shared__ float biasLds[144];        // b1[64], b2[64], b3[16 padded]

  const int tid  = threadIdx.x;
  const int lane = tid & 31;
  const int wave = tid >> 5;
  const int base = blockIdx.x * 128;
  const int p    = base + tid;
  const int N    = P.n;

  // ======================= hash-grid encoding =======================
  float x0 = P.x[3 * p + 0], x1 = P.x[3 * p + 1], x2 = P.x[3 * p + 2];
  float crd = P.cr[p] * 0.5f;

  const int myMt  = tid >> 4;
  const int myRow = tid & 15;
  _Float16* aBase = (_Float16*)&hFrag[myMt * 2 * 32];

  const float resf[LVLS] = {16.f, 34.f, 74.f, 161.f, 348.f, 752.f,
                            1625.f, 3511.f, 7584.f, 16384.f};

#pragma unroll
  for (int l = 0; l < LVLS; ++l) {
    float r  = resf[l];
    float s0 = x0 * r, s1 = x1 * r, s2 = x2 * r;
    float g0 = floorf(s0), g1 = floorf(s1), g2 = floorf(s2);
    unsigned u0 = (unsigned)(int)g0, u1 = (unsigned)(int)g1, u2 = (unsigned)(int)g2;
    float f0 = s0 - g0, f1 = s1 - g1, f2 = s2 - g2;

    vf4 acc = {0.f, 0.f, 0.f, 0.f};
    const vf4* tab = (const vf4*)(P.tables + (size_t)l * TSIZE * 4);
#pragma unroll
    for (int j = 0; j < 8; ++j) {
      unsigned c0 = (j >> 2) & 1u, c1 = (j >> 1) & 1u, c2 = j & 1u;
      unsigned hsh = (u0 + c0) ^ ((u1 + c1) * 2654435761u) ^ ((u2 + c2) * 805459861u);
      int idx = (int)(hsh & (unsigned)(TSIZE - 1));
      vf4 t = tab[idx];
      float w = (c0 ? f0 : 1.f - f0) * (c1 ? f1 : 1.f - f1) * (c2 ? f2 : 1.f - f2);
      acc += w * t;
    }
    // level downweight: approx_erf(1 / safe_sqrt(8 * l * crd))
    float den = sqrtf(fmaxf(8.f * (float)l * crd, 1e-8f));
    float ex  = 1.f / fmaxf(den, 1e-8f);
    float m   = sqrtf(fmaxf(1.f - __expf(-1.2732395447351627f * ex * ex), 0.f));
#pragma unroll
    for (int d = 0; d < 4; ++d)
      store_a_frag(aBase, myRow, d * LVLS + l, (_Float16)(acc[d] * m));
  }
#pragma unroll
  for (int k = 40; k < 64; ++k)
    store_a_frag(aBase, myRow, k, (_Float16)0.f);

  // ======================= 5 WMMA MLP heads =======================
  const int NOUT[5] = {1, 3, 3, 16, 4};
  const int hi  = lane >> 4;
  const int col = lane & 15;

  for (int hd = 0; hd < 5; ++hd) {
    __syncthreads();  // previous head compute done before restaging weights
    HeadW H = P.h[hd];
    const int nout = NOUT[hd];

    _Float16* bB = (_Float16*)bFrag;
    for (int i = tid; i < 64 * 64; i += 128) {   // W1: [40->64 K-pad][64]
      int k = i >> 6, n = i & 63;
      float v = (k < 40) ? H.W1[k * 64 + n] : 0.f;
      store_b_frag(bB, 0, k, n, (_Float16)v);
    }
    for (int i = tid; i < 64 * 64; i += 128) {   // W2: [64][64]
      int k = i >> 6, n = i & 63;
      store_b_frag(bB, 8, k, n, (_Float16)H.W2[k * 64 + n]);
    }
    for (int i = tid; i < 64 * 16; i += 128) {   // W3: [64][nout -> 16 N-pad]
      int k = i >> 4, n = i & 15;
      float v = (n < nout) ? H.W3[k * nout + n] : 0.f;
      store_b_frag(bB, 16, k, n, (_Float16)v);
    }
    if (tid < 64) { biasLds[tid] = H.b1[tid]; biasLds[64 + tid] = H.b2[tid]; }
    if (tid < 16) biasLds[128 + tid] = (tid < nout) ? H.b3[tid] : 0.f;
    __syncthreads();

#pragma unroll
    for (int mm = 0; mm < 2; ++mm) {
      const int mt = wave * 2 + mm;
      _Float16* actB = (_Float16*)&actFrag[mt * 2 * 32];

      // ---- layer 1: [16x64(K)] x [64x64] ----
      v16h a0 = hFrag[(mt * 2 + 0) * 32 + lane];
      v16h a1 = hFrag[(mt * 2 + 1) * 32 + lane];
      v8f c[4];
#pragma unroll
      for (int t = 0; t < 4; ++t) {
        v8f acc = {};
        acc = __builtin_amdgcn_wmma_f32_16x16x32_f16(false, a0, false,
              bFrag[(t * 2 + 0) * 32 + lane], (short)0, acc, false, false);
        acc = __builtin_amdgcn_wmma_f32_16x16x32_f16(false, a1, false,
              bFrag[(t * 2 + 1) * 32 + lane], (short)0, acc, false, false);
        c[t] = acc;
      }
#pragma unroll
      for (int t = 0; t < 4; ++t) {
        float b = biasLds[t * 16 + col];
#pragma unroll
        for (int v = 0; v < 8; ++v)
          store_a_frag(actB, v + 8 * hi, t * 16 + col, (_Float16)selu_f(c[t][v] + b));
      }

      // ---- layer 2: [16x64] x [64x64] ----
      a0 = actFrag[(mt * 2 + 0) * 32 + lane];
      a1 = actFrag[(mt * 2 + 1) * 32 + lane];
#pragma unroll
      for (int t = 0; t < 4; ++t) {
        v8f acc = {};
        acc = __builtin_amdgcn_wmma_f32_16x16x32_f16(false, a0, false,
              bFrag[(8 + t * 2 + 0) * 32 + lane], (short)0, acc, false, false);
        acc = __builtin_amdgcn_wmma_f32_16x16x32_f16(false, a1, false,
              bFrag[(8 + t * 2 + 1) * 32 + lane], (short)0, acc, false, false);
        c[t] = acc;
      }
#pragma unroll
      for (int t = 0; t < 4; ++t) {
        float b = biasLds[64 + t * 16 + col];
#pragma unroll
        for (int v = 0; v < 8; ++v)
          store_a_frag(actB, v + 8 * hi, t * 16 + col, (_Float16)selu_f(c[t][v] + b));
      }

      // ---- layer 3: [16x64] x [64x16] ----
      a0 = actFrag[(mt * 2 + 0) * 32 + lane];
      a1 = actFrag[(mt * 2 + 1) * 32 + lane];
      v8f o = {};
      o = __builtin_amdgcn_wmma_f32_16x16x32_f16(false, a0, false,
            bFrag[16 * 32 + lane], (short)0, o, false, false);
      o = __builtin_amdgcn_wmma_f32_16x16x32_f16(false, a1, false,
            bFrag[17 * 32 + lane], (short)0, o, false, false);

      float b3 = biasLds[128 + col];
#pragma unroll
      for (int v = 0; v < 8; ++v) {
        float val = o[v] + b3;
        long  pr  = (long)(base + mt * 16 + v + 8 * hi);
        if (hd == 0) {                       // density = exp(clip(sigma - 4))
          if (col == 0) {
            float z = fminf(fmaxf(val - 4.f, -15.f), 15.f);
            P.out[pr] = __expf(z);
          }
        } else if (hd == 1) {                // rgb + 0.5
          if (col < 3) P.out[(long)N + pr * 3 + col] = val + 0.5f;
        } else if (hd == 2) {                // gradient, row-normalized
          float q0 = __shfl(val, hi * 16 + 0, 32);
          float q1 = __shfl(val, hi * 16 + 1, 32);
          float q2 = __shfl(val, hi * 16 + 2, 32);
          float inv = 1.f / sqrtf(q0 * q0 + q1 * q1 + q2 * q2 + 1.f);
          if (col < 3) P.out[(long)4 * N + pr * 3 + col] = val * inv;
        } else if (hd == 3) {                // SH, fp16-quantized
          P.out[(long)7 * N + pr * 16 + col] = (float)(_Float16)val;
        } else {                             // attr
          if (col < 4) P.out[(long)23 * N + pr * 4 + col] = val;
        }
      }
    }
  }
}

extern "C" void kernel_launch(void* const* d_in, const int* in_sizes, int n_in,
                              void* d_out, int out_size, void* d_ws, size_t ws_size,
                              hipStream_t stream) {
  (void)n_in; (void)out_size; (void)d_ws; (void)ws_size;
  Params P;
  P.x      = (const float*)d_in[0];
  P.cr     = (const float*)d_in[1];
  P.tables = (const float*)d_in[2];
  for (int h = 0; h < 5; ++h) {
    P.h[h].W1 = (const float*)d_in[3 + h * 6 + 0];
    P.h[h].b1 = (const float*)d_in[3 + h * 6 + 1];
    P.h[h].W2 = (const float*)d_in[3 + h * 6 + 2];
    P.h[h].b2 = (const float*)d_in[3 + h * 6 + 3];
    P.h[h].W3 = (const float*)d_in[3 + h * 6 + 4];
    P.h[h].b3 = (const float*)d_in[3 + h * 6 + 5];
  }
  P.out = (float*)d_out;
  P.n   = in_sizes[1];               // cr has N elements

  int nBlocks = P.n / 128;           // N = 1048576 -> 8192 blocks
  ingp_fused_wmma<<<nBlocks, 128, 0, stream>>>(P);
}
